// GNNRegressor_30726196035937
// MI455X (gfx1250) — matrix operand order repaired and checked by reading, not compiled, not run on hardware
//
#include <hip/hip_runtime.h>
#include <hip/hip_bf16.h>
#include <stdint.h>

#define F 128            // feature width (F_IN == HID == 128)
#define LDSK 130         // LDS K-stride (65 dwords, odd -> conflict-free)

typedef __attribute__((ext_vector_type(16))) __bf16   v16bf;
typedef __attribute__((ext_vector_type(8)))  float    v8f;
typedef __attribute__((ext_vector_type(8)))  uint32_t v8u;

__device__ __forceinline__ __bf16 f2bf(float f) {
  // round-to-nearest-even f32 -> bf16
  uint32_t u = __builtin_bit_cast(uint32_t, f);
  u = u + 0x7FFFu + ((u >> 16) & 1u);
  return __builtin_bit_cast(__bf16, (uint16_t)(u >> 16));
}

__device__ __forceinline__ float lrelu(float x) { return x > 0.0f ? x : 0.01f * x; }

// ---------------------------------------------------------------------------
// GEMM: H[M x 128] = X[M x 128] @ W[128 x 128]  via v_wmma_f32_16x16x32_bf16
// Block = 256 threads (8 waves). Block tile = 128 rows x 128 cols.
// ---------------------------------------------------------------------------
__global__ __launch_bounds__(256) void gemm128_wmma(const float* __restrict__ X,
                                                    const float* __restrict__ W,
                                                    float* __restrict__ H, int M) {
  __shared__ __align__(16) __bf16 Xs[128 * LDSK];   // X tile, row-major [row][k]
  __shared__ __align__(16) __bf16 Wt[128 * LDSK];   // W transposed   [n][k]

  const int t    = threadIdx.x;
  const int row0 = blockIdx.x * 128;
  const bool full_tile = (row0 + 128 <= M);

  // Stage W^T as bf16 (coalesced global reads)
#pragma unroll
  for (int i = 0; i < 64; ++i) {
    int idx = t + i * 256;
    int k = idx >> 7, n = idx & 127;
    Wt[n * LDSK + k] = f2bf(W[idx]);
  }
  // Stage X tile as bf16 (zero-pad rows past M)
  if (full_tile) {
#pragma unroll
    for (int i = 0; i < 64; ++i) {
      int idx = t + i * 256;
      int r = idx >> 7, c = idx & 127;
      Xs[r * LDSK + c] = f2bf(X[(size_t)(row0 + r) * F + c]);
    }
  } else {
#pragma unroll
    for (int i = 0; i < 64; ++i) {
      int idx = t + i * 256;
      int r = idx >> 7, c = idx & 127;
      int gr = row0 + r;
      Xs[r * LDSK + c] = f2bf(gr < M ? X[(size_t)gr * F + c] : 0.0f);
    }
  }
  __syncthreads();

  const int lane = t & 31;
  const int wave = t >> 5;           // 0..7 : 16-column strip
  const int half = lane >> 4;        // 0/1
  const int rr   = lane & 15;
  const int colg = wave * 16 + rr;   // B / D column this lane owns

  // Preload the 4 B fragments (K = 4 x 32). B in this code's contiguous k order (A matches):
  // VGPR v holds packed K = half*16 + 2v, +1 for column N = lane&15.
  v8u bu[4];
#pragma unroll
  for (int kb = 0; kb < 4; ++kb) {
#pragma unroll
    for (int v = 0; v < 8; ++v) {
      int k = kb * 32 + half * 16 + 2 * v;
      bu[kb][v] = *reinterpret_cast<const uint32_t*>(&Wt[colg * LDSK + k]);
    }
  }

#pragma unroll
  for (int mt = 0; mt < 8; ++mt) {
    v8f acc = {};
    const int arow = mt * 16 + rr;   // A row (same row in both lane halves)
#pragma unroll
    for (int kb = 0; kb < 4; ++kb) {
      // A in the same contiguous k order as B: VGPR v holds packed K = half*16 + 2v, +1
      v8u au;
#pragma unroll
      for (int v = 0; v < 8; ++v) {
        int ka = kb * 32 + half * 16 + 2 * v;
        au[v] = *reinterpret_cast<const uint32_t*>(&Xs[arow * LDSK + ka]);
      }
      v16bf a = __builtin_bit_cast(v16bf, au);
      v16bf b = __builtin_bit_cast(v16bf, bu[kb]);
      acc = __builtin_amdgcn_wmma_f32_16x16x32_bf16(false, a, false, b,
                                                    (short)0, acc, false, false);
    }
    // D layout: VGPR r -> (M = r + half*8, N = lane&15)
    const int mrow = row0 + mt * 16 + half * 8;
    float* __restrict__ p = H + (size_t)mrow * F + colg;
    if (full_tile) {
      // unguarded fast path: immediate-offset stores, no exec-mask branches
#pragma unroll
      for (int r = 0; r < 8; ++r) p[(size_t)r * F] = acc[r];
    } else {
#pragma unroll
      for (int r = 0; r < 8; ++r) {
        if (mrow + r < M) p[(size_t)r * F] = acc[r];
      }
    }
  }
}

// ---------------------------------------------------------------------------
// Degree / normalization
// ---------------------------------------------------------------------------
__global__ void deg_init(float* __restrict__ deg, int N) {
  int i = blockIdx.x * blockDim.x + threadIdx.x;
  if (i < N) deg[i] = 1.0f;  // self-loop
}
__global__ void deg_edges(float* __restrict__ deg, const int* __restrict__ dst, int E) {
  int e = blockIdx.x * blockDim.x + threadIdx.x;
  if (e < E) atomicAdd(&deg[dst[e]], 1.0f);
}
__global__ void deg_finish(float* __restrict__ dinv, int N) {
  int i = blockIdx.x * blockDim.x + threadIdx.x;
  if (i < N) dinv[i] = rsqrtf(dinv[i]);  // deg >= 1 always
}

// ---------------------------------------------------------------------------
// Aggregation: out[i] = sum_{e:dst=i} h[src]*dinv[src]*dinv[i] + h[i]*dinv[i]^2
// ---------------------------------------------------------------------------
__global__ void agg_init(const float* __restrict__ h, const float* __restrict__ dinv,
                         float* __restrict__ out, int total) {
  int idx = blockIdx.x * blockDim.x + threadIdx.x;
  if (idx < total) {
    float di = dinv[idx >> 7];
    out[idx] = h[idx] * di * di;   // self-loop contribution, full overwrite
  }
}
__global__ void agg_edges(const float* __restrict__ h, const float* __restrict__ dinv,
                          const int* __restrict__ src, const int* __restrict__ dst,
                          float* __restrict__ out, int E) {
  int wid  = (int)((blockIdx.x * (unsigned)blockDim.x + threadIdx.x) >> 5);  // wave = edge
  int lane = threadIdx.x & 31;
  if (wid < E) {
    int s = src[wid], d = dst[wid];
    float w = dinv[s] * dinv[d];
    const float4 hv = *reinterpret_cast<const float4*>(h + (size_t)s * F + lane * 4);
    float* o = out + (size_t)d * F + lane * 4;
    atomicAdd(o + 0, hv.x * w);
    atomicAdd(o + 1, hv.y * w);
    atomicAdd(o + 2, hv.z * w);
    atomicAdd(o + 3, hv.w * w);
  }
}
__global__ void bias_act(float* __restrict__ buf, const float* __restrict__ bias, int total) {
  int idx = blockIdx.x * blockDim.x + threadIdx.x;
  if (idx < total) buf[idx] = lrelu(buf[idx] + bias[idx & 127]);
}

// ---------------------------------------------------------------------------
// Global mean pool + head MLP
// ---------------------------------------------------------------------------
__global__ void pool_zero(float* __restrict__ sums, float* __restrict__ cnt, int G) {
  int idx = blockIdx.x * blockDim.x + threadIdx.x;
  if (idx < G * F) sums[idx] = 0.0f;
  if (idx < G) cnt[idx] = 0.0f;
}
__global__ void pool_accum(const float* __restrict__ a, const int* __restrict__ batch,
                           float* __restrict__ sums, float* __restrict__ cnt, int N) {
  int wid  = (int)((blockIdx.x * (unsigned)blockDim.x + threadIdx.x) >> 5);  // wave = node
  int lane = threadIdx.x & 31;
  if (wid < N) {
    int b = batch[wid];
    const float4 v = *reinterpret_cast<const float4*>(a + (size_t)wid * F + lane * 4);
    float* o = sums + (size_t)b * F + lane * 4;
    atomicAdd(o + 0, v.x);
    atomicAdd(o + 1, v.y);
    atomicAdd(o + 2, v.z);
    atomicAdd(o + 3, v.w);
    if (lane == 0) atomicAdd(&cnt[b], 1.0f);
  }
}
__global__ __launch_bounds__(128) void mlp_head(const float* __restrict__ sums,
                                                const float* __restrict__ cnt,
                                                const float* __restrict__ W3,
                                                const float* __restrict__ b3,
                                                const float* __restrict__ W4,
                                                const float* __restrict__ b4,
                                                float* __restrict__ out) {
  __shared__ float mean[F];
  __shared__ float z[64];
  int g = blockIdx.x, t = threadIdx.x;
  float c = fmaxf(cnt[g], 1.0f);
  mean[t] = sums[(size_t)g * F + t] / c;
  __syncthreads();
  if (t < 64) {
    float acc = b3[t];
#pragma unroll 8
    for (int k = 0; k < F; ++k) acc += mean[k] * W3[k * 64 + t];
    z[t] = lrelu(acc) * W4[t];
  }
  __syncthreads();
  if (t == 0) {
    float s = b4[0];
#pragma unroll
    for (int j = 0; j < 64; ++j) s += z[j];
    out[g] = s;
  }
}

// ---------------------------------------------------------------------------
extern "C" void kernel_launch(void* const* d_in, const int* in_sizes, int n_in,
                              void* d_out, int out_size, void* d_ws, size_t ws_size,
                              hipStream_t stream) {
  const float* x     = (const float*)d_in[0];
  const int*   ei    = (const int*)  d_in[1];
  const int*   batch = (const int*)  d_in[2];
  const float* W1 = (const float*)d_in[3];  const float* b1 = (const float*)d_in[4];
  const float* W2 = (const float*)d_in[5];  const float* b2 = (const float*)d_in[6];
  const float* W3 = (const float*)d_in[7];  const float* b3 = (const float*)d_in[8];
  const float* W4 = (const float*)d_in[9];  const float* b4 = (const float*)d_in[10];

  const int N = in_sizes[0] / F;
  const int E = in_sizes[1] / 2;
  const int G = out_size;                    // OUT == 1
  const int* src = ei;
  const int* dst = ei + E;

  // workspace layout (16B aligned slices)
  auto align16 = [](size_t v) { return (v + 15) & ~(size_t)15; };
  char*  ws   = (char*)d_ws;
  size_t off  = 0;
  float* dinv = (float*)(ws + off); off = align16(off + (size_t)N * 4);
  float* P1   = (float*)(ws + off); off = align16(off + (size_t)N * F * 4);
  float* P2   = (float*)(ws + off); off = align16(off + (size_t)N * F * 4);
  float* sums = (float*)(ws + off); off = align16(off + (size_t)G * F * 4);
  float* cnt  = (float*)(ws + off);

  const int TB = 256;
  const int total = N * F;
  dim3 blkN((N + TB - 1) / TB), blkE((E + TB - 1) / TB), blkT((total + TB - 1) / TB);
  dim3 blkEW((E * 32 + TB - 1) / TB);   // wave-per-edge
  dim3 blkNW((N * 32 + TB - 1) / TB);   // wave-per-node
  dim3 gemmG((N + 127) / 128);

  // degree -> dinv (reinit every call: graph-replay safe)
  deg_init  <<<blkN, TB, 0, stream>>>(dinv, N);
  deg_edges <<<blkE, TB, 0, stream>>>(dinv, dst, E);
  deg_finish<<<blkN, TB, 0, stream>>>(dinv, N);

  // layer 1
  gemm128_wmma<<<gemmG, TB, 0, stream>>>(x, W1, P1, N);
  agg_init    <<<blkT, TB, 0, stream>>>(P1, dinv, P2, total);
  agg_edges   <<<blkEW, TB, 0, stream>>>(P1, dinv, src, dst, P2, E);
  bias_act    <<<blkT, TB, 0, stream>>>(P2, b1, total);

  // layer 2
  gemm128_wmma<<<gemmG, TB, 0, stream>>>(P2, W2, P1, N);
  agg_init    <<<blkT, TB, 0, stream>>>(P1, dinv, P2, total);
  agg_edges   <<<blkEW, TB, 0, stream>>>(P1, dinv, src, dst, P2, E);
  bias_act    <<<blkT, TB, 0, stream>>>(P2, b2, total);

  // pool + head
  pool_zero <<<(G * F + TB - 1) / TB, TB, 0, stream>>>(sums, cnt, G);
  pool_accum<<<blkNW, TB, 0, stream>>>(P2, batch, sums, cnt, N);
  mlp_head  <<<G, 128, 0, stream>>>(sums, cnt, W3, b3, W4, b4, (float*)d_out);
}
